// SlidingWindowHead_1382979469659
// MI455X (gfx1250) — matrix operand order repaired and checked by reading, not compiled, not run on hardware
//
#include <hip/hip_runtime.h>
#include <stdint.h>

typedef __attribute__((ext_vector_type(16))) __bf16 v16bf;
typedef __attribute__((ext_vector_type(2)))  __bf16 bf2;
typedef __attribute__((ext_vector_type(8)))  float  v8f;
typedef __attribute__((ext_vector_type(8)))  unsigned int v8u;

#define BDIM 4
#define TDIM 4096
#define CDIM 1024
#define HDIM 64
#define WIN  256

// ---- native bf16 conversion helpers (lower to v_cvt_pk_bf16_f32) ----
static __device__ __forceinline__ unsigned int pk2(float lo, float hi) {
  bf2 h = { (__bf16)lo, (__bf16)hi };
  return __builtin_bit_cast(unsigned int, h);
}
static __device__ __forceinline__ unsigned short bf1(float f) {
  __bf16 h = (__bf16)f;
  return __builtin_bit_cast(unsigned short, h);
}
static __device__ __forceinline__ v16bf cvt16(float4 a, float4 b, float4 c, float4 d) {
  v8u u = { pk2(a.x,a.y), pk2(a.z,a.w), pk2(b.x,b.y), pk2(b.z,b.w),
            pk2(c.x,c.y), pk2(c.z,c.w), pk2(d.x,d.y), pk2(d.z,d.w) };
  return __builtin_bit_cast(v16bf, u);
}
static __device__ __forceinline__ v8u load32bf(const unsigned short* p) {
  uint4 w0 = *(const uint4*)(p);
  uint4 w1 = *(const uint4*)(p + 8);
  v8u u = { w0.x, w0.y, w0.z, w0.w, w1.x, w1.y, w1.z, w1.w };
  return u;
}

// ---------------- one-shot weight conversion: Wq|Wk|Wv -> bf16 [3][64][1024] ----------------
__global__ __launch_bounds__(256) void cvtw_kernel(
    const float* __restrict__ Wq, const float* __restrict__ Wk,
    const float* __restrict__ Wv, unsigned short* __restrict__ Wb)
{
  const int N = HDIM * CDIM;                       // 65536 per matrix
  int tid = blockIdx.x * blockDim.x + threadIdx.x; // 3*N/8 threads, 8 elems each
  int m   = tid / (N / 8);
  int r   = (tid % (N / 8)) * 8;
  const float* src = (m == 0) ? Wq : (m == 1) ? Wk : Wv;
  float4 a = *(const float4*)(src + r);
  float4 b = *(const float4*)(src + r + 4);
  uint4 o = { pk2(a.x,a.y), pk2(a.z,a.w), pk2(b.x,b.y), pk2(b.z,b.w) };
  *(uint4*)(Wb + (size_t)m * N + r) = o;
}

// ---------------- fused projection: x[16xC] -> Q,K (row-major bf16) and V^T ----------------
// 8 waves/block (2 waves/EU -> 256-VGPR budget). All 12 B-fragments batch-loaded
// per K-step; sched_group_barrier pins loads ahead of the WMMA chain.
__global__ __launch_bounds__(256, 2) void proj_kernel(
    const float* __restrict__ x, const unsigned short* __restrict__ Wb,
    unsigned short* __restrict__ Qb, unsigned short* __restrict__ Kb,
    unsigned short* __restrict__ Vtb)
{
  const int lane = threadIdx.x & 31;
  const int wid  = threadIdx.x >> 5;
  const int half = lane >> 4;
  const int lm   = lane & 15;
  const int m0   = (blockIdx.x * 8 + wid) * 16;

  const float* xrow = x + (size_t)(m0 + lm) * CDIM;
  // B fragment f (= m*4+t): rows f*16+lm, contiguous K chunk
  const unsigned short* wbase = Wb + (size_t)lm * CDIM + half * 16;

  v8f acc[12] = {};

  // prologue: stage A floats for k0 = 0
  float4 a0 = *(const float4*)(xrow + half*16);
  float4 a1 = *(const float4*)(xrow + half*16 + 4);
  float4 a2 = *(const float4*)(xrow + half*16 + 8);
  float4 a3 = *(const float4*)(xrow + half*16 + 12);

  for (int k0 = 0; k0 < CDIM; k0 += 32) {
    // batch all 12 B-fragment loads first (24 x b128)
    v8u bfrag[12];
#pragma unroll
    for (int f = 0; f < 12; ++f)
      bfrag[f] = load32bf(wbase + (size_t)f * 16 * CDIM + k0);

    v16bf afrag = cvt16(a0, a1, a2, a3);            // uses last iteration's loads

    {  // issue next iteration's A loads now; consumed next iteration
      const int kn = (k0 + 32 < CDIM) ? (k0 + 32) : 0;
      const float* ap = xrow + kn + half * 16;
      if (k0 + 512 < CDIM)
        __builtin_prefetch((const void*)(ap + 512), 0, 3);  // global_prefetch_b8
      a0 = *(const float4*)(ap);
      a1 = *(const float4*)(ap + 4);
      a2 = *(const float4*)(ap + 8);
      a3 = *(const float4*)(ap + 12);
    }

#pragma unroll
    for (int f = 0; f < 12; ++f)
      acc[f] = __builtin_amdgcn_wmma_f32_16x16x32_bf16(
          false, afrag, false, __builtin_bit_cast(v16bf, bfrag[f]),
          (short)0, acc[f], false, false);

    // Scheduling pipeline for this loop body: all VMEM reads first (24 B + 4 A),
    // then the 12-WMMA chain. Forces distinct live buffers -> graduated waits.
    __builtin_amdgcn_sched_group_barrier(0x020, 28, 0);  // VMEM read
    __builtin_amdgcn_sched_group_barrier(0x008, 12, 0);  // MFMA/WMMA
  }

  // Q (f 0..3), K (f 4..7): row-major [B*T,64] bf16
#pragma unroll
  for (int t = 0; t < 4; ++t)
#pragma unroll
    for (int r = 0; r < 8; ++r) {
      const size_t o = (size_t)(m0 + r + 8*half) * HDIM + t*16 + lm;
      Qb[o] = bf1(acc[t][r]);
      Kb[o] = bf1(acc[4 + t][r]);
    }

  // V^T (f 8..11): [B][64][T]; 8 consecutive rows per lane -> one b128 store
  const int b  = m0 / TDIM;
  const int t0 = m0 % TDIM;
#pragma unroll
  for (int t = 0; t < 4; ++t) {
    uint4 ph;
    ph.x = pk2(acc[8 + t][0], acc[8 + t][1]);
    ph.y = pk2(acc[8 + t][2], acc[8 + t][3]);
    ph.z = pk2(acc[8 + t][4], acc[8 + t][5]);
    ph.w = pk2(acc[8 + t][6], acc[8 + t][7]);
    *(uint4*)(Vtb + (size_t)(b*HDIM + t*16 + lm) * TDIM + t0 + 8*half) = ph;
  }
}

// ---------------- sliding-window attention, flash-style ----------------
// 8 waves/block, one 16-query tile per wave.
__global__ __launch_bounds__(256) void attn_kernel(
    const unsigned short* __restrict__ Q,   // [B*T,64] bf16
    const unsigned short* __restrict__ K,   // [B*T,64] bf16
    const unsigned short* __restrict__ Vt,  // [B,64,T] bf16
    float* __restrict__ out)                // [B*T,64] f32
{
  __shared__ __align__(16) unsigned short pbuf[8 * 512];  // 1KB per wave

  const int lane = threadIdx.x & 31;
  const int wid  = threadIdx.x >> 5;
  const int half = lane >> 4;
  const int lm   = lane & 15;
  const int i0g  = (blockIdx.x * 8 + wid) * 16;
  const int b    = i0g / TDIM;
  const int i0   = i0g % TDIM;

  unsigned short* pb = pbuf + wid * 512;

  const unsigned short* qrow = Q + (size_t)(i0g + lm) * HDIM;
  v16bf qfrag[2];
#pragma unroll
  for (int c = 0; c < 2; ++c)
    qfrag[c] = __builtin_bit_cast(v16bf, load32bf(qrow + c*32 + half*16));

  float mrow[8], srow[8];
#pragma unroll
  for (int r = 0; r < 8; ++r) { mrow[r] = -1e30f; srow[r] = 0.0f; }
  v8f oacc[4] = {};

  const float scale = 0.03125f;  // C^-0.5 = 1/32

  for (int p = 0; p < 9; ++p) {
    const int jp = i0 - WIN + 32 * p;   // 32-key pair; covers [i0-256, i0+31]
    if (jp + 32 <= 0) continue;

    // batch all 4 K-fragment loads, then the 4 S-WMMAs
    v8u kf[4];
#pragma unroll
    for (int tj = 0; tj < 2; ++tj) {
      int jr = jp + 16*tj + lm;
      jr = jr < 0 ? 0 : (jr > TDIM - 1 ? TDIM - 1 : jr);
      const unsigned short* krow = K + (size_t)(b * TDIM + jr) * HDIM;
      kf[tj*2 + 0] = load32bf(krow + half*16);
      kf[tj*2 + 1] = load32bf(krow + 32 + half*16);
    }

    float sv[2][8];
#pragma unroll
    for (int tj = 0; tj < 2; ++tj) {
      v8f s = {};
#pragma unroll
      for (int c = 0; c < 2; ++c)
        s = __builtin_amdgcn_wmma_f32_16x16x32_bf16(
            false, qfrag[c], false, __builtin_bit_cast(v16bf, kf[tj*2 + c]),
            (short)0, s, false, false);
#pragma unroll
      for (int r = 0; r < 8; ++r) {
        const int i = i0 + r + 8*half;
        const int j = jp + 16*tj + lm;
        const bool valid = (j >= 0) && (j <= i) && (i - j < WIN);
        sv[tj][r] = valid ? s[r] * scale : -1e30f;
      }
    }

#pragma unroll
    for (int r = 0; r < 8; ++r) {
      float t = fmaxf(sv[0][r], sv[1][r]);
      t = fmaxf(t, __shfl_xor(t, 1, 32));
      t = fmaxf(t, __shfl_xor(t, 2, 32));
      t = fmaxf(t, __shfl_xor(t, 4, 32));
      t = fmaxf(t, __shfl_xor(t, 8, 32));
      const float mnew = fmaxf(mrow[r], t);
      const float corr = __expf(mrow[r] - mnew);
      mrow[r] = mnew;
      const float p0 = (sv[0][r] > -1e29f) ? __expf(sv[0][r] - mnew) : 0.0f;
      const float p1 = (sv[1][r] > -1e29f) ? __expf(sv[1][r] - mnew) : 0.0f;
      float rs = p0 + p1;
      rs += __shfl_xor(rs, 1, 32);
      rs += __shfl_xor(rs, 2, 32);
      rs += __shfl_xor(rs, 4, 32);
      rs += __shfl_xor(rs, 8, 32);
      srow[r] = srow[r] * corr + rs;
#pragma unroll
      for (int t4 = 0; t4 < 4; ++t4) oacc[t4][r] *= corr;
      pb[(r + 8*half) * 32 + lm]      = bf1(p0);
      pb[(r + 8*half) * 32 + 16 + lm] = bf1(p1);
    }

    // issue V-fragment loads before the LDS wait so they overlap the P transpose
    int jv = jp + half * 16;
    jv = jv < 0 ? 0 : (jv > TDIM - 16 ? TDIM - 16 : jv);
    v8u vf[4];
#pragma unroll
    for (int t4 = 0; t4 < 4; ++t4)
      vf[t4] = load32bf(Vt + (size_t)(b*HDIM + t4*16 + lm) * TDIM + jv);

    asm volatile("s_wait_dscnt 0" ::: "memory");
    const unsigned short* pr = pb + lm*32 + half*16;
    uint4 plo = *(const uint4*)(pr);
    uint4 phi = *(const uint4*)(pr + 8);
    v8u up = { plo.x, plo.y, plo.z, plo.w, phi.x, phi.y, phi.z, phi.w };
    v16bf pfrag = __builtin_bit_cast(v16bf, up);

#pragma unroll
    for (int t4 = 0; t4 < 4; ++t4)
      oacc[t4] = __builtin_amdgcn_wmma_f32_16x16x32_bf16(
          false, pfrag, false, __builtin_bit_cast(v16bf, vf[t4]),
          (short)0, oacc[t4], false, false);
  }

#pragma unroll
  for (int t4 = 0; t4 < 4; ++t4)
#pragma unroll
    for (int r = 0; r < 8; ++r)
      out[(size_t)(i0g + r + 8*half) * HDIM + t4*16 + lm] = oacc[t4][r] / srow[r];
}

extern "C" void kernel_launch(void* const* d_in, const int* in_sizes, int n_in,
                              void* d_out, int out_size, void* d_ws, size_t ws_size,
                              hipStream_t stream)
{
  const float* x  = (const float*)d_in[0];
  const float* Wq = (const float*)d_in[1];
  const float* Wk = (const float*)d_in[2];
  const float* Wv = (const float*)d_in[3];

  const size_t BT = (size_t)BDIM * TDIM;           // 16384 rows
  unsigned short* Qb  = (unsigned short*)d_ws;     // [B*T,64] bf16  (2 MB)
  unsigned short* Kb  = Qb + BT * HDIM;            // [B*T,64] bf16  (2 MB)
  unsigned short* Vtb = Kb + BT * HDIM;            // [B,64,T] bf16  (2 MB)
  unsigned short* Wb  = Vtb + BT * HDIM;           // [3,64,1024] bf16 (384 KB)

  const int blocks = (int)(BT / 16 / 8);           // 128 blocks x 8 waves

  cvtw_kernel<<<(3 * HDIM * CDIM / 8) / 256, 256, 0, stream>>>(Wq, Wk, Wv, Wb);
  proj_kernel<<<blocks, 256, 0, stream>>>(x, Wb, Qb, Kb, Vtb);
  attn_kernel<<<blocks, 256, 0, stream>>>(Qb, Kb, Vtb, (float*)d_out);
}
